// EnhancedGCNWithAttention_53455162966029
// MI455X (gfx1250) — matrix operand order repaired and checked by reading, not compiled, not run on hardware
//
#include <hip/hip_runtime.h>
#include <hip/hip_bf16.h>
#include <math.h>

// ---------------- problem constants (match reference) ----------------
#define N_NODES 4096
#define F_INP   768
#define HDIM    512
#define HEADS   4
#define DHEAD   128
#define CLS     32
#define NEDGE   131072

// ---------------- CDNA5 WMMA types ----------------
typedef __attribute__((ext_vector_type(16))) __bf16 v16bf;
typedef __attribute__((ext_vector_type(8)))  float  v8f;
typedef __attribute__((ext_vector_type(4)))  int    v4i;

union V16U { v16bf v; unsigned int u[8]; };

__device__ __forceinline__ unsigned short f32_to_bf16_r(float f) {
  unsigned int u = __float_as_uint(f);
  u += 0x7FFFu + ((u >> 16) & 1u);        // round-to-nearest-even
  return (unsigned short)(u >> 16);
}
__device__ __forceinline__ unsigned int packbf2(float a, float b) {
  return (unsigned int)f32_to_bf16_r(a) | ((unsigned int)f32_to_bf16_r(b) << 16);
}

// ---------------- CDNA5 async global->LDS path (ASYNCcnt) ----------------
#if __has_builtin(__builtin_amdgcn_global_load_async_to_lds_b128)
#define HAVE_ASYNC_LDS 1
#define ASYNC_CPY_B128(gsrc, ldst)                                        \
  __builtin_amdgcn_global_load_async_to_lds_b128(                         \
      (__attribute__((address_space(1))) v4i*)(gsrc),                     \
      (__attribute__((address_space(3))) v4i*)(ldst), 0, 0)
#else
#define HAVE_ASYNC_LDS 0
#endif

__device__ __forceinline__ void wait_async_lds() {
#if __has_builtin(__builtin_amdgcn_s_wait_asynccnt)
  __builtin_amdgcn_s_wait_asynccnt(0);
#endif
}

// =====================================================================
// Elementwise cast / transpose-cast (weights & activations -> bf16)
// =====================================================================
__global__ void cast_bf16_kernel(const float* __restrict__ a,
                                 unsigned short* __restrict__ o, size_t n) {
  size_t i = (size_t)blockIdx.x * blockDim.x + threadIdx.x;
  if (i < n) o[i] = f32_to_bf16_r(a[i]);
}

// a[R][C] -> o[C][R]  (bf16)
__global__ void transpose_cast_bf16_kernel(const float* __restrict__ a,
                                           unsigned short* __restrict__ o,
                                           int R, int C) {
  size_t i = (size_t)blockIdx.x * blockDim.x + threadIdx.x;
  if (i >= (size_t)R * C) return;
  int r = (int)(i / C), c = (int)(i % C);
  o[(size_t)c * R + r] = f32_to_bf16_r(a[i]);
}

// =====================================================================
// Degree / symmetric norm:  deg = 1 (self loop) + count(dst);  dinv = rsqrt
// =====================================================================
__global__ void deg_init_kernel(float* __restrict__ deg) {
  int i = blockIdx.x * blockDim.x + threadIdx.x;
  if (i < N_NODES) deg[i] = 1.0f;
}
__global__ void deg_count_kernel(const long long* __restrict__ dst,
                                 float* __restrict__ deg) {
  int e = blockIdx.x * blockDim.x + threadIdx.x;
  if (e < NEDGE) atomicAdd(&deg[(int)dst[e]], 1.0f);
}
__global__ void deg_rsqrt_kernel(float* __restrict__ deg) {
  int i = blockIdx.x * blockDim.x + threadIdx.x;
  if (i < N_NODES) deg[i] = rsqrtf(deg[i]);
}

// =====================================================================
// GCN aggregation: agg[n] = dinv[n]^2 * h[n]  (self loop seed)
//                  then per-edge atomic scatter of dinv[s]*dinv[d]*h[s]
// =====================================================================
__global__ void selfloop_init_kernel(const float* __restrict__ h,
                                     const float* __restrict__ dinv,
                                     float* __restrict__ agg) {
  size_t i = (size_t)blockIdx.x * blockDim.x + threadIdx.x;
  if (i >= (size_t)N_NODES * HDIM) return;
  int n = (int)(i >> 9);                   // / 512
  float w = dinv[n] * dinv[n];
  agg[i] = h[i] * w;
}

__global__ void edge_scatter_kernel(const float* __restrict__ h,
                                    const long long* __restrict__ src,
                                    const long long* __restrict__ dst,
                                    const float* __restrict__ dinv,
                                    float* __restrict__ agg) {
  int e = blockIdx.x * (blockDim.x >> 5) + (threadIdx.x >> 5);   // wave per edge
  if (e >= NEDGE) return;
  int lane = threadIdx.x & 31;
  int s = (int)src[e], d = (int)dst[e];
  float w = dinv[s] * dinv[d];
  const float* hs = h + (size_t)s * HDIM;
  float* ad = agg + (size_t)d * HDIM;
  #pragma unroll
  for (int f = lane; f < HDIM; f += 32)
    atomicAdd(&ad[f], hs[f] * w);
}

// out = relu(agg + b), plus bf16 copy for the next GEMM
__global__ void bias_relu_cast_kernel(const float* __restrict__ agg,
                                      const float* __restrict__ b,
                                      float* __restrict__ out,
                                      unsigned short* __restrict__ obf) {
  size_t i = (size_t)blockIdx.x * blockDim.x + threadIdx.x;
  if (i >= (size_t)N_NODES * HDIM) return;
  int f = (int)(i & (HDIM - 1));
  float v = fmaxf(agg[i] + b[f], 0.0f);
  out[i] = v;
  obf[i] = f32_to_bf16_r(v);
}

// =====================================================================
// bf16 WMMA GEMM:  C[M x Nn] = A[M x K] @ Bt[Nn x K]^T (+bias)(+residual)(relu?)
// Block = 4 waves sharing one 16-row A panel; panel is staged into LDS
// once per K-step via GLOBAL_LOAD_ASYNC_TO_LDS_B128 (ASYNCcnt), double-
// buffered so the async DMA for panel k+1 overlaps the WMMA on panel k.
// Waves consume A via ds_load, B via global b128 loads, and accumulate
// with v_wmma_f32_16x16x32_bf16.
// =====================================================================
#define GEMM_WPB 4   // waves (16-wide N tiles) per block

// stage one 16x32-bf16 A panel (1KB) into LDS: 64 x 16B chunks, 2 per lane
__device__ __forceinline__ void stage_a_panel(const unsigned short* __restrict__ Apan,
                                              int K, unsigned short* lds, int lane) {
  #pragma unroll
  for (int h = 0; h < 2; ++h) {
    int c = lane + 32 * h;                   // chunk id 0..63
    int row = c >> 2;                        // 4 chunks per 32-elem row
    int col = (c & 3) * 8;                   // 8 ushorts = 16 bytes
    const unsigned short* g = Apan + (size_t)row * K + col;
    unsigned short* l = lds + row * 32 + col;
#if HAVE_ASYNC_LDS
    ASYNC_CPY_B128(g, l);
#else
    *(uint4*)l = *(const uint4*)g;           // fallback: vmem load + ds_store_b128
#endif
  }
}

__global__ __launch_bounds__(32 * GEMM_WPB)
void wmma_gemm_kernel(const unsigned short* __restrict__ A,
                      const unsigned short* __restrict__ Bt,
                      const float* __restrict__ bias,
                      const float* __restrict__ residual,
                      float* __restrict__ C,
                      unsigned short* __restrict__ Cbf,
                      int M, int Nn, int K, int relu) {
  __shared__ unsigned short Apan[2][16 * 32];   // double-buffered A panel

  const int wave = threadIdx.x >> 5;
  const int lane = threadIdx.x & 31;
  const int half = lane >> 4;
  const int l16  = lane & 15;
  const int tileN = (blockIdx.x * GEMM_WPB + wave) * 16;
  const int tileM = blockIdx.y * 16;
  const bool active = tileN < Nn;               // inactive waves still barrier

  v8f acc = {};
  const unsigned short* Abase = A + (size_t)tileM * K;
  const unsigned short* Brow  = Bt + (size_t)((active ? tileN : 0) + l16) * K;

  // prologue: stage panel 0
  if (wave == 0) {
    stage_a_panel(Abase, K, Apan[0], lane);
    wait_async_lds();
  }
  __syncthreads();

  int buf = 0;
  for (int k0 = 0; k0 < K; k0 += 32) {
    const bool more = (k0 + 32) < K;
    if (more && wave == 0)                       // async DMA next panel, overlapped
      stage_a_panel(Abase + k0 + 32, K, Apan[buf ^ 1], lane);
    if (more) {
      __builtin_prefetch(Brow + k0 + 32, 0, 1);  // global_prefetch_b8 next B panel
    }

    V16U a, b;
    #pragma unroll
    for (int j = 0; j < 8; ++j) {
      // A 16x32 layout: lanes 0-15 K=0..7/16..23, lanes 16-31 K=8..15/24..31
      int ka = ((j < 4) ? (2 * j) : (16 + 2 * (j - 4))) + 8 * half;
      a.u[j] = *(const unsigned int*)&Apan[buf][l16 * 32 + ka];   // ds_load
      // B 32x16 layout: lanes 0-15 K=0..7/16..23, lanes 16-31 K=8..15/24..31 (2 K per VGPR)
      int kb = ((j < 4) ? (2 * j) : (16 + 2 * (j - 4))) + 8 * half;
      b.u[j] = *(const unsigned int*)(Brow + k0 + kb);
    }
    acc = __builtin_amdgcn_wmma_f32_16x16x32_bf16(false, a.v, false, b.v,
                                                  (short)0, acc, false, false);

    if (more) {
      if (wave == 0) wait_async_lds();           // next panel landed in LDS
      __syncthreads();                           // and everyone done with 'buf'
      buf ^= 1;
    }
  }

  if (!active) return;                           // no barriers past this point
  // C/D layout: VGPR i -> row i (lanes 0-15) / row 8+i (lanes 16-31), col = lane&15
  int n = tileN + l16;
  float bv = bias ? bias[n] : 0.0f;
  #pragma unroll
  for (int i = 0; i < 8; ++i) {
    int m = tileM + i + 8 * half;
    float v = acc[i] + bv;
    if (residual) v += residual[(size_t)m * Nn + n];
    if (relu) v = fmaxf(v, 0.0f);
    C[(size_t)m * Nn + n] = v;
    if (Cbf) Cbf[(size_t)m * Nn + n] = f32_to_bf16_r(v);
  }
}

// =====================================================================
// Flash attention: one wave per (head, 16-query tile). Q resident in
// registers (A layout, pre-scaled); stream keys 32 at a time:
//   S = Q Kt  (8 WMMAs), online softmax (intra-half shfl_xor reductions),
//   P -> LDS -> A-layout, O += P V (8 WMMAs).  Scores never hit HBM.
// =====================================================================
__global__ __launch_bounds__(32 * HEADS)
void flash_attn_kernel(const float* __restrict__ qkv,   // [N x 3*HDIM]
                       float* __restrict__ out) {       // [N x HDIM]
  __shared__ unsigned short pls[HEADS][16][32];         // P re-layout bounce

  const int head = threadIdx.x >> 5;                    // 4 waves = 4 heads
  const int lane = threadIdx.x & 31;
  const int half = lane >> 4;
  const int l16  = lane & 15;
  const int q0   = blockIdx.x * 16;
  const float scale = 0.08838834764831845f;             // 1/sqrt(128)

  // ---- load Q tile (16 x 128) into A-layout registers, pre-scaled ----
  V16U qa[4];
  const float* Qb = qkv + (size_t)(q0 + l16) * (3 * HDIM) + head * DHEAD;
  #pragma unroll
  for (int c = 0; c < 4; ++c) {
    #pragma unroll
    for (int j = 0; j < 8; ++j) {
      int ka = ((j < 4) ? (2 * j) : (16 + 2 * (j - 4))) + 8 * half + 32 * c;
      qa[c].u[j] = packbf2(Qb[ka] * scale, Qb[ka + 1] * scale);
    }
  }

  v8f o[8]; float mrow[8], lrow[8];
  #pragma unroll
  for (int t = 0; t < 8; ++t) { o[t] = (v8f){}; }
  #pragma unroll
  for (int i = 0; i < 8; ++i) { mrow[i] = -1e30f; lrow[i] = 0.0f; }

  for (int kt = 0; kt < N_NODES; kt += 32) {
    // ---- S = Q @ K^T for two 16-key groups ----
    v8f s0 = {}, s1 = {};
    const float* K0 = qkv + (size_t)(kt + l16) * (3 * HDIM) + HDIM + head * DHEAD;
    const float* K1 = K0 + (size_t)16 * (3 * HDIM);
    #pragma unroll
    for (int c = 0; c < 4; ++c) {
      V16U b0, b1;
      #pragma unroll
      for (int j = 0; j < 8; ++j) {
        int kb = 32 * c + ((j < 4) ? (2 * j) : (16 + 2 * (j - 4))) + 8 * half;   // B 32x16 layout over dh chunk
        b0.u[j] = packbf2(K0[kb], K0[kb + 1]);
        b1.u[j] = packbf2(K1[kb], K1[kb + 1]);
      }
      s0 = __builtin_amdgcn_wmma_f32_16x16x32_bf16(false, qa[c].v, false, b0.v,
                                                   (short)0, s0, false, false);
      s1 = __builtin_amdgcn_wmma_f32_16x16x32_bf16(false, qa[c].v, false, b1.v,
                                                   (short)0, s1, false, false);
    }

    // ---- online softmax (rows live in 16-lane halves) ----
    #pragma unroll
    for (int i = 0; i < 8; ++i) {
      float rm = fmaxf(s0[i], s1[i]);
      #pragma unroll
      for (int off = 1; off < 16; off <<= 1) rm = fmaxf(rm, __shfl_xor(rm, off, 32));
      float mnew = fmaxf(mrow[i], rm);
      float p0 = __expf(s0[i] - mnew);
      float p1 = __expf(s1[i] - mnew);
      float rs = p0 + p1;
      #pragma unroll
      for (int off = 1; off < 16; off <<= 1) rs += __shfl_xor(rs, off, 32);
      float alpha = __expf(mrow[i] - mnew);
      lrow[i] = lrow[i] * alpha + rs;
      mrow[i] = mnew;
      #pragma unroll
      for (int t = 0; t < 8; ++t) o[t][i] *= alpha;
      s0[i] = p0; s1[i] = p1;                 // keep P in the S registers
    }

    // ---- re-layout P (C/D layout -> A layout) through LDS ----
    #pragma unroll
    for (int i = 0; i < 8; ++i) {
      pls[head][i + 8 * half][l16]      = f32_to_bf16_r(s0[i]);
      pls[head][i + 8 * half][16 + l16] = f32_to_bf16_r(s1[i]);
    }
    __syncthreads();
    V16U pa;
    #pragma unroll
    for (int j = 0; j < 8; ++j) {
      int kk = 16 * half + 2 * j;
      pa.u[j] = *(const unsigned int*)&pls[head][l16][kk];
    }
    __syncthreads();

    // ---- O += P @ V  (8 dh column tiles) ----
    const float* Vb = qkv + (size_t)(kt + 16 * half) * (3 * HDIM)
                    + 2 * HDIM + head * DHEAD;
    #pragma unroll
    for (int t = 0; t < 8; ++t) {
      V16U vb;
      const float* Vc = Vb + t * 16 + l16;
      #pragma unroll
      for (int j = 0; j < 8; ++j)
        vb.u[j] = packbf2(Vc[(size_t)(2 * j) * (3 * HDIM)],
                          Vc[(size_t)(2 * j + 1) * (3 * HDIM)]);
      o[t] = __builtin_amdgcn_wmma_f32_16x16x32_bf16(false, pa.v, false, vb.v,
                                                     (short)0, o[t], false, false);
    }
  }

  // ---- finalize: O /= l, write [q0+row, head*128 + t*16 + n] ----
  #pragma unroll
  for (int i = 0; i < 8; ++i) {
    float inv = 1.0f / lrow[i];
    int m = q0 + i + 8 * half;
    #pragma unroll
    for (int t = 0; t < 8; ++t)
      out[(size_t)m * HDIM + head * DHEAD + t * 16 + l16] = o[t][i] * inv;
  }
}

// =====================================================================
// Orchestration
// =====================================================================
static inline dim3 gemm_grid(int M, int Nn) {
  int ntiles = (Nn / 16 + GEMM_WPB - 1) / GEMM_WPB;
  return dim3(ntiles, M / 16);
}

extern "C" void kernel_launch(void* const* d_in, const int* in_sizes, int n_in,
                              void* d_out, int out_size, void* d_ws, size_t ws_size,
                              hipStream_t stream) {
  const float*     x     = (const float*)d_in[0];
  const long long* edge  = (const long long*)d_in[1];
  const float*     W1    = (const float*)d_in[2];
  const float*     b1    = (const float*)d_in[3];
  const float*     in_w  = (const float*)d_in[4];
  const float*     in_b  = (const float*)d_in[5];
  const float*     out_w = (const float*)d_in[6];
  const float*     out_b = (const float*)d_in[7];
  const float*     W2    = (const float*)d_in[8];
  const float*     b2    = (const float*)d_in[9];
  const float*     Wc    = (const float*)d_in[10];
  const float*     bc    = (const float*)d_in[11];
  const long long* src = edge;
  const long long* dst = edge + NEDGE;

  // bump allocator over scratch
  char* ws = (char*)d_ws; size_t off = 0;
  auto alloc = [&](size_t bytes) -> void* {
    void* p = ws + off; off = (off + bytes + 255) & ~(size_t)255; return p;
  };
  unsigned short* x_bf    = (unsigned short*)alloc((size_t)N_NODES * F_INP * 2);
  unsigned short* W1t_bf  = (unsigned short*)alloc((size_t)HDIM * F_INP * 2);
  unsigned short* inw_bf  = (unsigned short*)alloc((size_t)3 * HDIM * HDIM * 2);
  unsigned short* outw_bf = (unsigned short*)alloc((size_t)HDIM * HDIM * 2);
  unsigned short* W2t_bf  = (unsigned short*)alloc((size_t)HDIM * HDIM * 2);
  unsigned short* Wct_bf  = (unsigned short*)alloc((size_t)CLS * HDIM * 2);
  float* dinv   = (float*)alloc((size_t)N_NODES * 4);
  float* f1     = (float*)alloc((size_t)N_NODES * HDIM * 4);   // xw1 / xw2
  float* agg    = (float*)alloc((size_t)N_NODES * HDIM * 4);
  float* h1     = (float*)alloc((size_t)N_NODES * HDIM * 4);
  unsigned short* h1_bf = (unsigned short*)alloc((size_t)N_NODES * HDIM * 2);
  float* qkv    = (float*)alloc((size_t)N_NODES * 3 * HDIM * 4);
  float* attno  = (float*)alloc((size_t)N_NODES * HDIM * 4);
  unsigned short* at_bf = (unsigned short*)alloc((size_t)N_NODES * HDIM * 2);
  float* h2     = (float*)alloc((size_t)N_NODES * HDIM * 4);
  unsigned short* h2_bf = (unsigned short*)alloc((size_t)N_NODES * HDIM * 2);
  float* h3     = (float*)alloc((size_t)N_NODES * HDIM * 4);
  unsigned short* h3_bf = (unsigned short*)alloc((size_t)N_NODES * HDIM * 2);

  const int TB = 256;
  auto g1 = [&](size_t n) { return (unsigned)((n + TB - 1) / TB); };

  // ---- precision staging: weights/activations -> bf16 (transposed where needed)
  cast_bf16_kernel<<<g1((size_t)N_NODES * F_INP), TB, 0, stream>>>(x, x_bf, (size_t)N_NODES * F_INP);
  transpose_cast_bf16_kernel<<<g1((size_t)F_INP * HDIM), TB, 0, stream>>>(W1, W1t_bf, F_INP, HDIM);
  cast_bf16_kernel<<<g1((size_t)3 * HDIM * HDIM), TB, 0, stream>>>(in_w, inw_bf, (size_t)3 * HDIM * HDIM);
  cast_bf16_kernel<<<g1((size_t)HDIM * HDIM), TB, 0, stream>>>(out_w, outw_bf, (size_t)HDIM * HDIM);
  transpose_cast_bf16_kernel<<<g1((size_t)HDIM * HDIM), TB, 0, stream>>>(W2, W2t_bf, HDIM, HDIM);
  transpose_cast_bf16_kernel<<<g1((size_t)HDIM * CLS), TB, 0, stream>>>(Wc, Wct_bf, HDIM, CLS);

  // ---- degrees / symmetric norm
  deg_init_kernel<<<g1(N_NODES), TB, 0, stream>>>(dinv);
  deg_count_kernel<<<g1(NEDGE), TB, 0, stream>>>(dst, dinv);
  deg_rsqrt_kernel<<<g1(N_NODES), TB, 0, stream>>>(dinv);

  // ---- GCN layer 1: xw1 = x @ W1 ; aggregate ; relu(+b1)
  wmma_gemm_kernel<<<gemm_grid(N_NODES, HDIM), 32 * GEMM_WPB, 0, stream>>>(
      x_bf, W1t_bf, nullptr, nullptr, f1, nullptr, N_NODES, HDIM, F_INP, 0);
  selfloop_init_kernel<<<g1((size_t)N_NODES * HDIM), TB, 0, stream>>>(f1, dinv, agg);
  edge_scatter_kernel<<<(NEDGE + 7) / 8, TB, 0, stream>>>(f1, src, dst, dinv, agg);
  bias_relu_cast_kernel<<<g1((size_t)N_NODES * HDIM), TB, 0, stream>>>(agg, b1, h1, h1_bf);

  // ---- attention: qkv = h1 @ in_w^T + in_b ; flash attention ; out proj + residual
  wmma_gemm_kernel<<<gemm_grid(N_NODES, 3 * HDIM), 32 * GEMM_WPB, 0, stream>>>(
      h1_bf, inw_bf, in_b, nullptr, qkv, nullptr, N_NODES, 3 * HDIM, HDIM, 0);
  flash_attn_kernel<<<N_NODES / 16, 32 * HEADS, 0, stream>>>(qkv, attno);
  cast_bf16_kernel<<<g1((size_t)N_NODES * HDIM), TB, 0, stream>>>(attno, at_bf, (size_t)N_NODES * HDIM);
  wmma_gemm_kernel<<<gemm_grid(N_NODES, HDIM), 32 * GEMM_WPB, 0, stream>>>(
      at_bf, outw_bf, out_b, h1, h2, h2_bf, N_NODES, HDIM, HDIM, 0);

  // ---- GCN layer 2: xw2 = h2 @ W2 ; aggregate ; relu(+b2)
  wmma_gemm_kernel<<<gemm_grid(N_NODES, HDIM), 32 * GEMM_WPB, 0, stream>>>(
      h2_bf, W2t_bf, nullptr, nullptr, f1, nullptr, N_NODES, HDIM, HDIM, 0);
  selfloop_init_kernel<<<g1((size_t)N_NODES * HDIM), TB, 0, stream>>>(f1, dinv, agg);
  edge_scatter_kernel<<<(NEDGE + 7) / 8, TB, 0, stream>>>(f1, src, dst, dinv, agg);
  bias_relu_cast_kernel<<<g1((size_t)N_NODES * HDIM), TB, 0, stream>>>(agg, b2, h3, h3_bf);

  // ---- classifier: out = h3 @ Wc + bc
  wmma_gemm_kernel<<<gemm_grid(N_NODES, CLS), 32 * GEMM_WPB, 0, stream>>>(
      h3_bf, Wct_bf, bc, nullptr, (float*)d_out, nullptr, N_NODES, CLS, HDIM, 0);
}
